// GORU_17179869457
// MI455X (gfx1250) — compile-verified, matrix operand order checked by reading, not run on hardware
//
#include <hip/hip_runtime.h>
#include <hip/hip_bf16.h>

// GORU scan for MI455X (gfx1250): bf16 WMMA 16x16x32, fp32 accumulate.
// B=128, T=2048, D_IN=128, H=256.
// Grid: 8 WGs x 512 threads (16 waves, wave32). Wave w owns H-cols [16w,16w+16).
//
// On-chip residency plan (256-VGPR/wave cap observed in round 1 asm):
//   Registers : Uh B-frags (64 VGPR) + Wz/Wr/Wx B-frags (96 VGPR) = 160 persistent
//   LDS       : Rz+Rr B-frags pre-swizzled (256 KB) + h tile bf16 (8.4 KB)
// Per step each wave reloads Rz/Rr frags with 2x ds_load_b128 per fragment.

typedef __attribute__((ext_vector_type(16))) __bf16 v16bf;
typedef __attribute__((ext_vector_type(8)))  float  v8f;

#define T_STEPS 2048
#define D_IN    128
#define HDIM    256
#define LDH     264           // padded bf16 row stride (528B): 4-bank row skew
#define RFRAG_BYTES (16 * 2 * 8 * 1024)              // 256 KB: [wave][mat][chunk] slots
#define HSM_OFF     RFRAG_BYTES
#define SMEM_BYTES  (RFRAG_BYTES + 16 * LDH * 2)     // + 8448 B h tile

static __device__ __forceinline__ v8f wmma_bf16(v16bf a, v16bf b, v8f c) {
    return __builtin_amdgcn_wmma_f32_16x16x32_bf16(
        false, a, false, b, (short)0, c, false, false);
}

static __device__ __forceinline__ float sigmoidf_fast(float v) {
    return 1.0f / (1.0f + __expf(-v));
}

// Gather a bf16 B-fragment (32-K x 16-N tile) from row-major fp32 M[k*HDIM+n].
// B layout (mirror of A, lane->N):
//   lanes 0-15 : N=lane,    elems 0-7 = K kb+0..7,  elems 8-15 = K kb+16..23
//   lanes 16-31: N=lane-16, elems 0-7 = K kb+8..15, elems 8-15 = K kb+24..31
static __device__ __forceinline__ v16bf load_bfrag(const float* M, int kbase,
                                                   int khalf, int n_glob) {
    v16bf f;
    const int kb = kbase + khalf;
#pragma unroll
    for (int e = 0; e < 8; ++e)
        f[e] = (__bf16)M[(kb + e) * HDIM + n_glob];
#pragma unroll
    for (int e = 0; e < 8; ++e)
        f[8 + e] = (__bf16)M[(kb + 16 + e) * HDIM + n_glob];
    return f;
}

__global__ __launch_bounds__(512, 1)
void goru_scan_kernel(const float* __restrict__ x,
                      const float* __restrict__ h0,
                      const float* __restrict__ Wz,
                      const float* __restrict__ Wr,
                      const float* __restrict__ Wx,
                      const float* __restrict__ Uh,
                      const float* __restrict__ Rz,
                      const float* __restrict__ Rr,
                      const float* __restrict__ Bz,
                      const float* __restrict__ Br,
                      const float* __restrict__ Bh,
                      float* __restrict__ out) {
    extern __shared__ char smem[];
    char*   rfrag = smem;                               // 256 KB frag slots
    __bf16* hsm   = (__bf16*)(smem + HSM_OFF);          // h tile [16][LDH]

    const int lane    = threadIdx.x & 31;
    const int wv      = threadIdx.x >> 5;               // 0..15
    const int nloc    = lane & 15;
    const int khalf   = (lane >> 4) * 8;                // 0 / 8
    const int n_glob  = wv * 16 + nloc;                 // this lane's H column
    const int rowbase = blockIdx.x * 16;                // batch tile

    // Slot base for (mat, chunk): each slot = 2 x 512B half-arrays, lane*16 stride.
    char* wslot = rfrag + wv * (2 * 8 * 1024);

    // ---- One-time: stage Rz, Rr fragments into LDS (pre-swizzled) ----
#pragma unroll
    for (int mat = 0; mat < 2; ++mat) {
        const float* M = mat ? Rr : Rz;
#pragma unroll
        for (int c = 0; c < 8; ++c) {
            v16bf f = load_bfrag(M, c * 32, khalf, n_glob);
            char* base = wslot + (mat * 8 + c) * 1024;
            *reinterpret_cast<uint4*>(base + lane * 16)       = reinterpret_cast<uint4*>(&f)[0];
            *reinterpret_cast<uint4*>(base + 512 + lane * 16) = reinterpret_cast<uint4*>(&f)[1];
        }
    }

    // ---- One-time: Uh + W fragments stay in registers (160 VGPRs) ----
    v16bf rBu[8];
#pragma unroll
    for (int c = 0; c < 8; ++c) rBu[c] = load_bfrag(Uh, c * 32, khalf, n_glob);
    v16bf wBz[4], wBr[4], wBx[4];
#pragma unroll
    for (int c = 0; c < 4; ++c) {
        wBz[c] = load_bfrag(Wz, c * 32, khalf, n_glob);
        wBr[c] = load_bfrag(Wr, c * 32, khalf, n_glob);
        wBx[c] = load_bfrag(Wx, c * 32, khalf, n_glob);
    }

    const float bz = Bz[n_glob];
    const float br = Br[n_glob];
    const float bh = Bh[n_glob];

    // ---- Init h (C/D layout: VGPR i -> row i+khalf, col n_glob) ----
    v8f h;
#pragma unroll
    for (int i = 0; i < 8; ++i) {
        const int m = i + khalf;
        const float v = h0[(size_t)(rowbase + m) * HDIM + n_glob];
        h[i] = v;
        hsm[m * LDH + n_glob] = (__bf16)v;
    }
    __syncthreads();

    const float* xrow0 = x + (size_t)(rowbase + nloc) * T_STEPS * D_IN;

    for (int t = 0; t < T_STEPS; ++t) {
        v8f az = {}, ar = {}, au = {}, ax = {};

        // ---- Input projections: x_t[16x128] @ {Wz,Wr,Wx} (A from global) ----
        const float* xrow = xrow0 + (size_t)t * D_IN;
#pragma unroll
        for (int c = 0; c < 4; ++c) {
            const int kb = c * 32 + khalf;
            const float4 p0 = *reinterpret_cast<const float4*>(xrow + kb);
            const float4 p1 = *reinterpret_cast<const float4*>(xrow + kb + 4);
            const float4 p2 = *reinterpret_cast<const float4*>(xrow + kb + 16);
            const float4 p3 = *reinterpret_cast<const float4*>(xrow + kb + 20);
            v16bf a;
            a[0]  = (__bf16)p0.x; a[1]  = (__bf16)p0.y; a[2]  = (__bf16)p0.z; a[3]  = (__bf16)p0.w;
            a[4]  = (__bf16)p1.x; a[5]  = (__bf16)p1.y; a[6]  = (__bf16)p1.z; a[7]  = (__bf16)p1.w;
            a[8]  = (__bf16)p2.x; a[9]  = (__bf16)p2.y; a[10] = (__bf16)p2.z; a[11] = (__bf16)p2.w;
            a[12] = (__bf16)p3.x; a[13] = (__bf16)p3.y; a[14] = (__bf16)p3.z; a[15] = (__bf16)p3.w;
            az = wmma_bf16(a, wBz[c], az);
            ar = wmma_bf16(a, wBr[c], ar);
            ax = wmma_bf16(a, wBx[c], ax);
        }

        // ---- Recurrent: h[16x256] @ {Rz,Rr,Uh} ----
#pragma unroll
        for (int c = 0; c < 8; ++c) {
            // h A-fragment from LDS
            const int kb = c * 32 + khalf;
            const __bf16* hp = &hsm[nloc * LDH + kb];
            v16bf a;
            reinterpret_cast<uint4*>(&a)[0] = *reinterpret_cast<const uint4*>(hp);
            reinterpret_cast<uint4*>(&a)[1] = *reinterpret_cast<const uint4*>(hp + 16);

            // Rz B-fragment from LDS slot
            {
                const char* base = wslot + c * 1024;
                v16bf bzf;
                reinterpret_cast<uint4*>(&bzf)[0] = *reinterpret_cast<const uint4*>(base + lane * 16);
                reinterpret_cast<uint4*>(&bzf)[1] = *reinterpret_cast<const uint4*>(base + 512 + lane * 16);
                az = wmma_bf16(a, bzf, az);
            }
            // Rr B-fragment from LDS slot
            {
                const char* base = wslot + (8 + c) * 1024;
                v16bf brf;
                reinterpret_cast<uint4*>(&brf)[0] = *reinterpret_cast<const uint4*>(base + lane * 16);
                reinterpret_cast<uint4*>(&brf)[1] = *reinterpret_cast<const uint4*>(base + 512 + lane * 16);
                ar = wmma_bf16(a, brf, ar);
            }
            // Uh from registers
            au = wmma_bf16(a, rBu[c], au);
        }
        __syncthreads();  // all hsm reads for this step complete

        // ---- Gate math in C/D layout; h_new -> LDS (bf16) + out (fp32) ----
#pragma unroll
        for (int i = 0; i < 8; ++i) {
            const float zt = sigmoidf_fast(az[i] + bz);
            const float rt = sigmoidf_fast(ar[i] + br);
            const float zi = ax[i] + rt * au[i];
            const float aa = fabsf(zi);
            const float g  = copysignf(fmaxf(aa + bh, 0.0f), zi);  // modReLU
            const float hn = zt * h[i] + (1.0f - zt) * g;
            h[i] = hn;
            const int m = i + khalf;
            hsm[m * LDH + n_glob] = (__bf16)hn;
            out[((size_t)(rowbase + m) * T_STEPS + t) * HDIM + n_glob] = hn;
        }
        __syncthreads();  // hsm updated before next step's reads
    }
}

extern "C" void kernel_launch(void* const* d_in, const int* in_sizes, int n_in,
                              void* d_out, int out_size, void* d_ws, size_t ws_size,
                              hipStream_t stream) {
    (void)in_sizes; (void)n_in; (void)d_ws; (void)ws_size; (void)out_size;
    const float* x  = (const float*)d_in[0];
    const float* h0 = (const float*)d_in[1];
    const float* Wz = (const float*)d_in[2];
    const float* Wr = (const float*)d_in[3];
    const float* Wx = (const float*)d_in[4];
    const float* Uh = (const float*)d_in[5];
    const float* Rz = (const float*)d_in[6];
    const float* Rr = (const float*)d_in[7];
    const float* Bz = (const float*)d_in[8];
    const float* Br = (const float*)d_in[9];
    const float* Bh = (const float*)d_in[10];
    float* out = (float*)d_out;

    hipLaunchKernelGGL(goru_scan_kernel, dim3(128 / 16), dim3(512), SMEM_BYTES, stream,
                       x, h0, Wz, Wr, Wx, Uh, Rz, Rr, Bz, Br, Bh, out);
}
